// EfficientAttention3d_9483287790341
// MI455X (gfx1250) — compile-verified, matrix operand order checked by reading, not run on hardware
//
#include <hip/hip_runtime.h>
#include <hip/hip_bf16.h>

// ---------------------------------------------------------------------------
// EfficientAttention3d, algebraically reduced (q-softmax over size-1 axis == 1):
//   S[b,h,c] = sum_n exp(key[b,h,n]) * x[b,c,n],  E[b,h] = sum_n exp(key)
//   ctx[b,h,v] = bv + Wv @ (S/E);  delta[b,c] = br + Wr @ ctx;  out = x + delta
// K1: WMMA reduce over x (one full read of x).  K2: tiny combine.  K3: stream add.
// ---------------------------------------------------------------------------

#define BATCH 4
#define CIN 64
#define HEADS 4
#define VAL 32
#define NTOK (64 * 64 * 64)          // 262144 tokens per batch
#define CHUNKS 256                    // blocks per batch in the reduce kernel
#define TPB_TOK (NTOK / CHUNKS)       // 1024 tokens per block
#define WAVES 8
#define TOK_PER_WAVE (TPB_TOK / WAVES) // 128
#define TILES (TOK_PER_WAVE / 32)      // 4 tiles of 32 tokens per wave

typedef __attribute__((ext_vector_type(16))) _Float16 v16h;
typedef __attribute__((ext_vector_type(8)))  float    v8f;

// K index inside a 16-bit 16x32 (A) / 32x16 (B) WMMA operand for
// v16h element e, lane l.  (ISA 7.12.2, 16-bit A-matrix layout.)
__device__ __forceinline__ int opK(int lane, int e) {
    int p = e >> 1, o = e & 1;
    return ((p & 4) ? 16 : 0) + ((lane & 16) ? 8 : 0) + ((p & 3) << 1) + o;
}

__global__ __launch_bounds__(256) void ea3d_reduce(
    const float* __restrict__ x, const float* __restrict__ Wk,
    const float* __restrict__ bk, float* __restrict__ wsS,
    float* __restrict__ wsE) {
    // per-wave LDS regions (pitches 34 halves = 17 dwords: conflict-free strides)
    __shared__ _Float16 xA[WAVES][CIN][34];    // x tile, [channel][token] f16
    __shared__ _Float16 wbuf[WAVES][16][34];   // exp weights, rows 4..15 stay 0
    __shared__ float    Sbuf[WAVES][HEADS][CIN];
    __shared__ float    Ebuf[WAVES][HEADS];

    const int lane = threadIdx.x & 31;
    const int wv   = threadIdx.x >> 5;
    const int b     = blockIdx.x / CHUNKS;
    const int chunk = blockIdx.x % CHUNKS;
    const long long baseTok = (long long)chunk * TPB_TOK + (long long)wv * TOK_PER_WAVE;
    const float* xb = x + (long long)b * CIN * NTOK;

    // zero wbuf (rows >=4 must remain zero: they are the padded WMMA rows)
    for (int i = threadIdx.x; i < WAVES * 16 * 34; i += 256)
        ((_Float16*)wbuf)[i] = (_Float16)0.f;
    __syncthreads();

    // Preload Wk into WMMA A-layout registers (2 K-steps over 64 input chans).
    v16h Ak0, Ak1;
    {
        int m = lane & 15;
        for (int e = 0; e < 16; ++e) {
            int K = opK(lane, e);
            Ak0[e] = (m < HEADS) ? (_Float16)Wk[m * CIN + K]      : (_Float16)0.f;
            Ak1[e] = (m < HEADS) ? (_Float16)Wk[m * CIN + 32 + K] : (_Float16)0.f;
        }
    }
    float bkv[HEADS];
    for (int h = 0; h < HEADS; ++h) bkv[h] = bk[h];

    v8f accS[4];
    for (int ct = 0; ct < 4; ++ct) accS[ct] = (v8f)(0.f);
    float Ep[HEADS] = {0.f, 0.f, 0.f, 0.f};

    for (int tile = 0; tile < TILES; ++tile) {
        const long long t0 = baseTok + tile * 32;
        // ---- stage 64ch x 32tok tile into LDS (coalesced b128 reads) ----
        {
            int cOff = lane >> 3;          // 0..3
            int t    = (lane & 7) * 4;     // 0..28
            for (int i = 0; i < 16; ++i) {
                int c = i * 4 + cOff;
                const float4 v =
                    *reinterpret_cast<const float4*>(xb + (long long)c * NTOK + t0 + t);
                xA[wv][c][t + 0] = (_Float16)v.x;
                xA[wv][c][t + 1] = (_Float16)v.y;
                xA[wv][c][t + 2] = (_Float16)v.z;
                xA[wv][c][t + 3] = (_Float16)v.w;
            }
        }
        __syncthreads();

        // ---- keys via WMMA: D(16h x 16tok) = Wk(16x64) @ x(64x16tok) ----
        for (int tt = 0; tt < 2; ++tt) {
            v16h Bk0, Bk1;
            int n = lane & 15;
            for (int e = 0; e < 16; ++e) {
                int K = opK(lane, e);
                Bk0[e] = xA[wv][K][tt * 16 + n];
                Bk1[e] = xA[wv][32 + K][tt * 16 + n];
            }
            v8f d = (v8f)(0.f);
            d = __builtin_amdgcn_wmma_f32_16x16x32_f16(false, Ak0, false, Bk0,
                                                       (short)0, d, false, false);
            d = __builtin_amdgcn_wmma_f32_16x16x32_f16(false, Ak1, false, Bk1,
                                                       (short)0, d, false, false);
            if (lane < 16) {  // lanes 0..15 hold rows M=0..7 -> heads 0..3
                for (int r = 0; r < HEADS; ++r) {
                    float wgt = __expf(d[r] + bkv[r]);   // keys are O(1): no max shift needed
                    Ep[r] += wgt;
                    wbuf[wv][r][tt * 16 + lane] = (_Float16)wgt;
                }
            }
        }
        __syncthreads();

        // ---- S via WMMA: D(16h x 16c) += w(16h x 32tok) @ x(32tok x 16c) ----
        v16h Aw;
        {
            int m = lane & 15;
            for (int e = 0; e < 16; ++e) Aw[e] = wbuf[wv][m][opK(lane, e)];
        }
        for (int ct = 0; ct < 4; ++ct) {
            v16h Bs;
            int n = lane & 15;
            for (int e = 0; e < 16; ++e) Bs[e] = xA[wv][ct * 16 + n][opK(lane, e)];
            accS[ct] = __builtin_amdgcn_wmma_f32_16x16x32_f16(
                false, Aw, false, Bs, (short)0, accS[ct], false, false);
        }
        __syncthreads();
    }

    // ---- wave-level epilogue: E reduce across lanes 0..15, spill S rows ----
    for (int h = 0; h < HEADS; ++h) {
        float e = Ep[h];
        e += __shfl_xor(e, 8, 16);
        e += __shfl_xor(e, 4, 16);
        e += __shfl_xor(e, 2, 16);
        e += __shfl_xor(e, 1, 16);
        if (lane == 0) Ebuf[wv][h] = e;
    }
    if (lane < 16) {
        for (int ct = 0; ct < 4; ++ct)
            for (int r = 0; r < HEADS; ++r)
                Sbuf[wv][r][ct * 16 + lane] = accS[ct][r];
    }
    __syncthreads();

    // ---- block-level combine (fixed order -> deterministic) ----
    {
        int t = threadIdx.x;  // t -> (h = t>>6, c = t&63)
        float s = 0.f;
        for (int w = 0; w < WAVES; ++w) s += Sbuf[w][t >> 6][t & 63];
        wsS[(long long)blockIdx.x * 256 + t] = s;
        if (t < HEADS) {
            float e = 0.f;
            for (int w = 0; w < WAVES; ++w) e += Ebuf[w][t];
            wsE[blockIdx.x * HEADS + t] = e;
        }
    }
}

__global__ __launch_bounds__(256) void ea3d_combine(
    const float* __restrict__ wsS, const float* __restrict__ wsE,
    const float* __restrict__ Wv, const float* __restrict__ bv,
    const float* __restrict__ Wr, const float* __restrict__ br,
    float* __restrict__ delta) {
    __shared__ float sS[HEADS][CIN];
    __shared__ float sE[HEADS];
    __shared__ float ctx[VAL];
    const int t = threadIdx.x;
    for (int b = 0; b < BATCH; ++b) {
        {   // sum per-block partials (fixed order)
            float s = 0.f;
            for (int k = 0; k < CHUNKS; ++k)
                s += wsS[((long long)(b * CHUNKS + k)) * 256 + t];
            sS[t >> 6][t & 63] = s;
        }
        if (t < HEADS) {
            float e = 0.f;
            for (int k = 0; k < CHUNKS; ++k) e += wsE[(b * CHUNKS + k) * HEADS + t];
            sE[t] = e;
        }
        __syncthreads();
        if (t < VAL) {  // ctx[hv] = bv + Wv @ (S/E)   (head = hv/8)
            int h = t >> 3;
            float inv = 1.0f / sE[h];
            float acc = bv[t];
            for (int c = 0; c < CIN; ++c) acc += Wv[t * CIN + c] * (sS[h][c] * inv);
            ctx[t] = acc;
        }
        __syncthreads();
        if (t < CIN) {  // delta[c] = br + Wr @ ctx
            float acc = br[t];
            for (int j = 0; j < VAL; ++j) acc += Wr[t * VAL + j] * ctx[j];
            delta[b * CIN + t] = acc;
        }
        __syncthreads();
    }
}

__global__ __launch_bounds__(256) void ea3d_apply(
    const float* __restrict__ x, const float* __restrict__ delta,
    float* __restrict__ out) {
    const long long Q = (long long)BATCH * CIN * NTOK / 4;  // float4 count
    const long long stride = (long long)gridDim.x * blockDim.x;
    for (long long i = (long long)blockIdx.x * 256 + threadIdx.x; i < Q; i += stride) {
        int bc = (int)(i >> 16);  // NTOK/4 = 65536 float4 per (b,c) plane
        float4 v = reinterpret_cast<const float4*>(x)[i];
        float d = delta[bc];
        v.x += d; v.y += d; v.z += d; v.w += d;
        reinterpret_cast<float4*>(out)[i] = v;
        __builtin_prefetch(reinterpret_cast<const char*>(x) + (i + stride) * 16, 0, 0);
    }
}

extern "C" void kernel_launch(void* const* d_in, const int* in_sizes, int n_in,
                              void* d_out, int out_size, void* d_ws, size_t ws_size,
                              hipStream_t stream) {
    const float* x  = (const float*)d_in[0];
    const float* Wk = (const float*)d_in[1];
    const float* bk = (const float*)d_in[2];
    // d_in[3] = Wq, d_in[4] = bq: dead inputs (softmax over a size-1 axis == 1)
    const float* Wv = (const float*)d_in[5];
    const float* bv = (const float*)d_in[6];
    const float* Wr = (const float*)d_in[7];
    const float* br = (const float*)d_in[8];
    float* out = (float*)d_out;

    float* wsS   = (float*)d_ws;                               // [B*CHUNKS][256]
    float* wsE   = wsS + (size_t)BATCH * CHUNKS * 256;         // [B*CHUNKS][4]
    float* delta = wsE + (size_t)BATCH * CHUNKS * HEADS;       // [B][64]

    ea3d_reduce<<<BATCH * CHUNKS, 256, 0, stream>>>(x, Wk, bk, wsS, wsE);
    ea3d_combine<<<1, 256, 0, stream>>>(wsS, wsE, Wv, bv, Wr, br, delta);
    ea3d_apply<<<8192, 256, 0, stream>>>(x, delta, out);
}
